// SDA_3977139716505
// MI455X (gfx1250) — compile-verified, matrix operand order checked by reading, not run on hardware
//
#include <hip/hip_runtime.h>
#include <hip/hip_bf16.h>
#include <stdint.h>

typedef __attribute__((ext_vector_type(16))) _Float16 v16h;
typedef __attribute__((ext_vector_type(8)))  float    v8f;
typedef _Float16 h16;

constexpr int C_    = 128;   // channels
constexpr int NIMG  = 8;     // T*B
constexpr int B_    = 2;
constexpr int T_    = 4;
constexpr int NSP   = 8192;  // 8*32*32 spatial per image
constexpr int H_    = 4;     // heads
constexpr int D_    = 32;    // C/H
#define THETA_ 0.7f
#define VTH_   1.0f
#define EPS_   1e-5f

// wait for this wave's outstanding async (LDS<-global) transfers
__device__ __forceinline__ void wait_async0() {
#if __has_builtin(__builtin_amdgcn_s_wait_asynccnt)
    __builtin_amdgcn_s_wait_asynccnt(0);
#else
    asm volatile("s_wait_asynccnt 0x0" ::: "memory");
#endif
}

// issue one 16-byte async global->LDS transfer (ASYNCcnt tracked)
__device__ __forceinline__ void async_g2l_b128(uint32_t lds_byte_addr, const void* gptr) {
    asm volatile("global_load_async_to_lds_b128 %0, %1, off"
                 :: "v"(lds_byte_addr), "v"((uint64_t)(uintptr_t)gptr)
                 : "memory");
}

// ---------------------------------------------------------------- cvt f32->f16
__global__ void k_cvt(const float* __restrict__ in, h16* __restrict__ out, int n) {
    int i = blockIdx.x * blockDim.x + threadIdx.x;
    if (i < n) out[i] = (h16)in[i];
}

// --------------------------------------------------- weight prep (fused taps)
__global__ void k_prep_w(const float* __restrict__ qw, const float* __restrict__ kw,
                         const float* __restrict__ pw, h16* __restrict__ wq,
                         h16* __restrict__ wk, h16* __restrict__ wp) {
    int i = blockIdx.x * blockDim.x + threadIdx.x;
    const int NQ = 27 * C_ * C_;
    if (i < NQ) {
        int tap = i / (C_ * C_);
        int rem = i % (C_ * C_);
        int co = rem >> 7, ci = rem & 127;
        const float* base = qw + (size_t)(co * C_ + ci) * 27;
        float val = base[tap];
        if (tap == 13) {                       // center tap: fuse -theta*kdiff
            float kd = 0.f;
#pragma unroll
            for (int j = 0; j < 9; ++j) kd += base[j] + base[18 + j];
            val -= THETA_ * kd;
        }
        wq[(size_t)tap * C_ * C_ + co * C_ + ci] = (h16)val;
    } else if (i < NQ + C_ * C_) {
        int j = i - NQ;
        wk[j] = (h16)kw[j];
    } else if (i < NQ + 2 * C_ * C_) {
        int j = i - NQ - C_ * C_;
        wp[j] = (h16)pw[j];
    }
}

// ------------------------------------------------- per-channel BN statistics
__global__ __launch_bounds__(256) void k_stats(const float* __restrict__ in,
                                               float* __restrict__ stats) {
    __shared__ float ssum[256], ssq[256];
    int c = blockIdx.x;
    float s = 0.f, q = 0.f;
    for (int img = 0; img < NIMG; ++img) {
        const float* p = in + ((size_t)img * C_ + c) * NSP;
        for (int n = threadIdx.x; n < NSP; n += 256) {
            float v = p[n];
            s += v; q += v * v;
        }
    }
    ssum[threadIdx.x] = s; ssq[threadIdx.x] = q;
    __syncthreads();
    for (int off = 128; off > 0; off >>= 1) {
        if ((int)threadIdx.x < off) {
            ssum[threadIdx.x] += ssum[threadIdx.x + off];
            ssq[threadIdx.x]  += ssq[threadIdx.x + off];
        }
        __syncthreads();
    }
    if (threadIdx.x == 0) {
        float cnt = (float)(NIMG * NSP);
        float mu  = ssum[0] / cnt;
        float var = ssq[0] / cnt - mu * mu;
        stats[2 * c]     = mu;
        stats[2 * c + 1] = rsqrtf(var + EPS_);
    }
}

// ------------------------------------------------------ implicit-GEMM conv
// A: [ntaps][128 co][128 ci] f16   B: [img][128 ci][8192] f16 (spikes)
// Out: [img][128 co][8192] f32.  conv3!=0 -> 27 shifted taps w/ zero pad.
// A tile staged via async global->LDS (ASYNCcnt); B tile via VGPR path
// (needs per-element zero-padding at the conv borders).
__global__ __launch_bounds__(256)
void k_gemm(const h16* __restrict__ A, const h16* __restrict__ Bm,
            float* __restrict__ Out, int conv3) {
    __shared__ h16 Alds[128][144];   // [M][K], 288B row stride (16B aligned)
    __shared__ h16 Blds[64][136];    // [N][K] K-contiguous, padded
    const int tid  = threadIdx.x;
    const int wave = tid >> 5;
    const int lane = tid & 31;
    const int img  = blockIdx.y;
    const int n0   = blockIdx.x * 64;
    const int ntaps = conv3 ? 27 : 1;
    const int wm = (wave & 3) * 32;  // wave M-base (4 waves along M)
    const int wn = (wave >> 2) * 32; // wave N-base (2 waves along N)
    const int r  = lane & 15;
    const int hi = lane >> 4;

    v8f acc[2][2] = {};

    for (int tap = 0; tap < ntaps; ++tap) {
        const int dz = conv3 ? (tap / 9) - 1 : 0;
        const int dy = conv3 ? ((tap / 3) % 3) - 1 : 0;
        const int dx = conv3 ? (tap % 3) - 1 : 0;

        // ---- A tile 128x128: async global->LDS, 8x b128 per thread
        const h16* Ap = A + (size_t)tap * C_ * C_;
#pragma unroll
        for (int i = 0; i < 8; ++i) {
            int u   = tid + i * 256;            // 16-byte unit id, 2048 total
            int co  = u >> 4;                   // 16 units per 128-half row
            int ci0 = (u & 15) * 8;
            uint32_t la = (uint32_t)(uintptr_t)&Alds[co][ci0];
            async_g2l_b128(la, Ap + (size_t)u * 8);
        }

        // ---- B tile 128ci x 64n with shift + zero pad, 32 halves per thread
#pragma unroll 4
        for (int i = 0; i < 32; ++i) {
            int flat = tid + i * 256;
            int ci = flat >> 6;
            int nn = flat & 63;
            int n  = n0 + nn;
            int z  = (n >> 10) + dz;
            int y  = ((n >> 5) & 31) + dy;
            int x  = (n & 31) + dx;
            h16 val = (h16)0.0f;
            if ((unsigned)z < 8u && (unsigned)y < 32u && (unsigned)x < 32u)
                val = Bm[((size_t)img * C_ + ci) * NSP + ((z * 32 + y) * 32 + x)];
            Blds[nn][ci] = val;
        }
        if (tap + 1 < ntaps)
            __builtin_prefetch(Bm + (size_t)img * C_ * NSP + n0, 0, 1);

        wait_async0();          // this wave's async LDS writes landed
        __syncthreads();        // all waves' A/B tile writes visible

#pragma unroll
        for (int k0 = 0; k0 < 128; k0 += 32) {
            v16h af[2], bf[2];
#pragma unroll
            for (int i = 0; i < 2; ++i) {
                // 16-bit A 16x32 layout: lane r/hi -> K groups [hi*8..] & [16+hi*8..]
                const h16* s0 = &Alds[wm + i * 16 + r][k0 + hi * 8];
#pragma unroll
                for (int e = 0; e < 8; ++e) { af[i][e] = s0[e]; af[i][8 + e] = s0[16 + e]; }
            }
#pragma unroll
            for (int j = 0; j < 2; ++j) {
                const h16* s0 = &Blds[wn + j * 16 + r][k0 + hi * 8];
#pragma unroll
                for (int e = 0; e < 8; ++e) { bf[j][e] = s0[e]; bf[j][8 + e] = s0[16 + e]; }
            }
#pragma unroll
            for (int i = 0; i < 2; ++i)
#pragma unroll
                for (int j = 0; j < 2; ++j)
                    acc[i][j] = __builtin_amdgcn_wmma_f32_16x16x32_f16(
                        false, af[i], false, bf[j], (short)0, acc[i][j], false, false);
        }
        __syncthreads();
    }

    // C/D layout: VGPR rr holds M=rr (lanes 0-15) / M=rr+8 (lanes 16-31), N=lane&15
#pragma unroll
    for (int i = 0; i < 2; ++i) {
#pragma unroll
        for (int j = 0; j < 2; ++j) {
            int nb = n0 + wn + j * 16 + r;
#pragma unroll
            for (int rr = 0; rr < 8; ++rr) {
                int row = wm + i * 16 + hi * 8 + rr;
                Out[((size_t)img * C_ + row) * NSP + nb] = acc[i][j][rr];
            }
        }
    }
}

// ---------------------------------------------------- fused BN + LIF over T
__global__ __launch_bounds__(256)
void k_lif(const float* __restrict__ in, const float* __restrict__ stats,
           const float* __restrict__ gamma, const float* __restrict__ beta,
           h16* __restrict__ spk) {
    int flat = blockIdx.x * 256 + threadIdx.x;     // over B*C*NSP
    int b = flat >> 20;
    int c = (flat >> 13) & 127;
    int n = flat & 8191;
    float mu = stats[2 * c], rstd = stats[2 * c + 1];
    float g  = gamma[c] * rstd;
    float bb = beta[c] - mu * g;                   // xn = val*g + bb
    float vm = 0.f;
    for (int t = 0; t < T_; ++t) {
        size_t idx = ((size_t)(t * B_ + b) * C_ + c) * NSP + n;
        float xn = in[idx] * g + bb;
        vm = 0.5f * (vm + xn);                     // v += (x - v)/tau, tau=2
        float s = (vm >= VTH_) ? 1.f : 0.f;
        spk[idx] = (h16)s;
        vm *= (1.f - s);                           // hard reset
    }
}

// ---------------------- spiking attention: lif(q.k) per head, then lif(a*v)
__global__ __launch_bounds__(256)
void k_attn(const h16* __restrict__ q, const h16* __restrict__ k,
            const h16* __restrict__ v, h16* __restrict__ out) {
    int flat = blockIdx.x * 256 + threadIdx.x;     // over B*H*NSP = 65536
    int b = flat >> 15;
    int h = (flat >> 13) & (H_ - 1);
    int n = flat & 8191;
    float va = 0.f;
    float vo[D_];
#pragma unroll
    for (int d = 0; d < D_; ++d) vo[d] = 0.f;
    for (int t = 0; t < T_; ++t) {
        size_t base = ((size_t)(t * B_ + b) * C_ + h * D_) * NSP + n;
        float dot = 0.f;
#pragma unroll
        for (int d = 0; d < D_; ++d)
            dot += (float)q[base + (size_t)d * NSP] * (float)k[base + (size_t)d * NSP];
        va = 0.5f * (va + dot);
        float a = (va >= VTH_) ? 1.f : 0.f;
        va *= (1.f - a);
#pragma unroll
        for (int d = 0; d < D_; ++d) {
            float o = a * (float)v[base + (size_t)d * NSP];
            vo[d] = 0.5f * (vo[d] + o);
            float s = (vo[d] >= VTH_) ? 1.f : 0.f;
            out[base + (size_t)d * NSP] = (h16)s;
            vo[d] *= (1.f - s);
        }
    }
}

// --------------------------------------------------------------- final BN
__global__ void k_bn_out(const float* __restrict__ in, const float* __restrict__ stats,
                         const float* __restrict__ gamma, const float* __restrict__ beta,
                         float* __restrict__ out, int n) {
    int i = blockIdx.x * blockDim.x + threadIdx.x;
    if (i >= n) return;
    int c = (i >> 13) & 127;
    out[i] = (in[i] - stats[2 * c]) * stats[2 * c + 1] * gamma[c] + beta[c];
}

// ---------------------------------------------------------------- launcher
extern "C" void kernel_launch(void* const* d_in, const int* in_sizes, int n_in,
                              void* d_out, int out_size, void* d_ws, size_t ws_size,
                              hipStream_t stream) {
    const float* x  = (const float*)d_in[0];
    const float* qw = (const float*)d_in[1];
    const float* qg = (const float*)d_in[2];
    const float* qb = (const float*)d_in[3];
    const float* kw = (const float*)d_in[4];
    const float* kg = (const float*)d_in[5];
    const float* kb = (const float*)d_in[6];
    const float* vg = (const float*)d_in[7];
    const float* vb = (const float*)d_in[8];
    const float* pw = (const float*)d_in[9];
    const float* pg = (const float*)d_in[10];
    const float* pb = (const float*)d_in[11];
    float* out = (float*)d_out;

    char* ws = (char*)d_ws;
    size_t off = 0;
    auto carve = [&](size_t bytes) -> void* {
        void* p = ws + off;
        off += (bytes + 255) & ~(size_t)255;
        return p;
    };
    const size_t NTOT = (size_t)NIMG * C_ * NSP;   // 8.39M
    h16*   xh  = (h16*)  carve(NTOT * 2);
    float* qn  = (float*)carve(NTOT * 4);          // reused for proj output
    float* kn  = (float*)carve(NTOT * 4);
    h16*   qs  = (h16*)  carve(NTOT * 2);
    h16*   ksp = (h16*)  carve(NTOT * 2);
    h16*   vsp = (h16*)  carve(NTOT * 2);
    h16*   ssp = (h16*)  carve(NTOT * 2);
    h16*   wqf = (h16*)  carve((size_t)27 * C_ * C_ * 2);
    h16*   wkf = (h16*)  carve((size_t)C_ * C_ * 2);
    h16*   wpf = (h16*)  carve((size_t)C_ * C_ * 2);
    float* stq = (float*)carve(2 * C_ * 4);
    float* stk = (float*)carve(2 * C_ * 4);
    float* stv = (float*)carve(2 * C_ * 4);
    float* stp = (float*)carve(2 * C_ * 4);

    k_cvt<<<(int)(NTOT / 256), 256, 0, stream>>>(x, xh, (int)NTOT);
    k_prep_w<<<(27 * C_ * C_ + 2 * C_ * C_ + 255) / 256, 256, 0, stream>>>(
        qw, kw, pw, wqf, wkf, wpf);
    k_stats<<<C_, 256, 0, stream>>>(x, stv);

    dim3 gg(NSP / 64, NIMG);
    k_gemm<<<gg, 256, 0, stream>>>(wqf, xh, qn, 1);   // fused 3x3x3 conv
    k_stats<<<C_, 256, 0, stream>>>(qn, stq);
    k_gemm<<<gg, 256, 0, stream>>>(wkf, xh, kn, 0);   // 1x1 conv (k)
    k_stats<<<C_, 256, 0, stream>>>(kn, stk);

    const int NLIF = B_ * C_ * NSP;                   // 2.10M
    k_lif<<<NLIF / 256, 256, 0, stream>>>(qn, stq, qg, qb, qs);
    k_lif<<<NLIF / 256, 256, 0, stream>>>(kn, stk, kg, kb, ksp);
    k_lif<<<NLIF / 256, 256, 0, stream>>>(x,  stv, vg, vb, vsp);

    k_attn<<<(B_ * H_ * NSP) / 256, 256, 0, stream>>>(qs, ksp, vsp, ssp);

    k_gemm<<<gg, 256, 0, stream>>>(wpf, ssp, qn, 0);  // 1x1 proj conv
    k_stats<<<C_, 256, 0, stream>>>(qn, stp);
    k_bn_out<<<(int)(NTOT / 256), 256, 0, stream>>>(qn, stp, pg, pb, out, (int)NTOT);
}
